// InrAwaSA_74947179316071
// MI455X (gfx1250) — matrix operand order repaired and checked by reading, not compiled, not run on hardware
//
#include <hip/hip_runtime.h>

typedef __attribute__((ext_vector_type(16))) _Float16 v16h;
typedef __attribute__((ext_vector_type(8)))  _Float16 v8h;
typedef __attribute__((ext_vector_type(4)))  _Float16 v4h;
typedef __attribute__((ext_vector_type(8)))  float    v8f;
typedef __attribute__((ext_vector_type(4)))  float    v4f;
typedef __attribute__((ext_vector_type(4)))  unsigned int v4u;
typedef __attribute__((ext_vector_type(8)))  int      v8i;
typedef __attribute__((ext_vector_type(4)))  int      v4i;

#define B_     4
#define N_     4096
#define D_     256
#define QTILE  128           // q rows per block = 8 waves * 16
#define KVTILE 32
#define KTS    (D_ + 8)      // sK row stride (halfs) -> 528B rows, conflict-free frags
#define VTS    (KVTILE + 8)  // sVt row stride (halfs)
#define PS     40            // sP row stride (halfs)
#define NTILES (N_ / KVTILE)
// 1/sqrt(256) * log2(e): softmax done in exp2 domain
#define QSCALE 0.0901684761f

static __device__ __forceinline__ v16h cat8(v8h a, v8h b) {
  v16h r;
#pragma unroll
  for (int i = 0; i < 8; ++i) { r[i] = a[i]; r[8 + i] = b[i]; }
  return r;
}

// ---------------- pre-pass: x fp32 -> f16 (row-major) into workspace ----------------
__global__ __launch_bounds__(256, 1)
void cvt_f16_kernel(const float* __restrict__ x, _Float16* __restrict__ xh) {
  const size_t i = (size_t)blockIdx.x * 2048 + (size_t)threadIdx.x * 8;
  v4f a = *(const v4f*)(x + i);
  v4f b = *(const v4f*)(x + i + 4);
  v8h h;
#pragma unroll
  for (int j = 0; j < 4; ++j) { h[j] = (_Float16)a[j]; h[4 + j] = (_Float16)b[j]; }
  *(v8h*)(xh + i) = h;
}

// ---------------- TDM descriptor issue: 32x256 f16 tile -> LDS (padded rows) --------
static __device__ __forceinline__ void tdm_issue(const _Float16* gtile, unsigned lds_off) {
  const unsigned long long ga = (unsigned long long)(size_t)gtile;
  v4u g0;
  g0[0] = 1u;                                   // count=1 (valid), user flags 0
  g0[1] = lds_off;                              // lds_addr (bytes)
  g0[2] = (unsigned)(ga & 0xffffffffu);         // global_addr[31:0]
  g0[3] = (unsigned)((ga >> 32) & 0x1ffffffu)   // global_addr[56:32]
        | (2u << 30);                           // type=2 ("image")
  v8i g1;
  g1[0] = (1 << 16)                             // data_size = 2 bytes
        | (1 << 20)                             // pad_enable
        | (6 << 22)                             // pad_interval: 128 DWORDs (one 512B row)
        | (3 << 25);                            // pad_amount: 4 DWORDs (16B) -> 264-half stride
  g1[1] = (int)(256u << 16);                    // tensor_dim0 = 256 (elements)
  g1[2] = (int)(4096u << 16);                   // tensor_dim1 = 4096 rows
  g1[3] = (int)(256u << 16);                    // tile_dim0 = 256
  g1[4] = 32;                                   // tile_dim1 = 32 rows
  g1[5] = 256;                                  // tensor_dim0_stride = 256 elements
  g1[6] = 0;
  g1[7] = 0;
  v4i gz = {0, 0, 0, 0};
#if __has_include(<hip/amd_detail/amd_gfx1250_TDM.h>)
  v8i gz8 = {0, 0, 0, 0, 0, 0, 0, 0};
  __builtin_amdgcn_tensor_load_to_lds(g0, g1, gz, gz, gz8, 0);
#else
  __builtin_amdgcn_tensor_load_to_lds(g0, g1, gz, gz, 0);
#endif
}

// ---------------- fused flash-attention, f16 source + TDM double buffering ----------
__global__ __launch_bounds__(256, 1)
void fa_wmma_tdm_kernel(const _Float16* __restrict__ xh, float* __restrict__ out) {
  __shared__ _Float16 sK [2][KVTILE * KTS];  // 2*16896 B, TDM destination
  __shared__ _Float16 sVt[D_ * VTS];         // 20480 B, transposed V tile
  __shared__ _Float16 sP [8 * 16 * PS];      // 10240 B, per-wave P scratch

  const int tid  = threadIdx.x;
  const int lane = tid & 31;
  const int wv   = tid >> 5;
  const int hh   = lane >> 4;
  const int l16  = lane & 15;

  const int    b     = blockIdx.x / (N_ / QTILE);
  const int    qt    = blockIdx.x % (N_ / QTILE);
  const size_t xb    = (size_t)b * N_ * D_;
  const int    qrow0 = qt * QTILE + wv * 16;
  const _Float16* xhb = xh + xb;

  // ---- Q tile -> registers in WMMA A-fragment layout, scale*log2e folded in ----
  v16h Qf[8];
  {
    const _Float16* qp = xhb + (size_t)(qrow0 + l16) * D_;
    const _Float16  sc = (_Float16)QSCALE;
#pragma unroll
    for (int c = 0; c < 8; ++c) {
      v8h lo = *(const v8h*)(qp + 32 * c + 16 * hh);
      v8h hi = *(const v8h*)(qp + 32 * c + 16 * hh + 8);
      v16h q;
#pragma unroll
      for (int i = 0; i < 8; ++i) { q[i] = lo[i] * sc; q[8 + i] = hi[i] * sc; }
      Qf[c] = q;
    }
  }

  v8f O[16];
#pragma unroll
  for (int c = 0; c < 16; ++c) { v8f z = {}; O[c] = z; }
  float mrow[8], lrow[8];
#pragma unroll
  for (int r = 0; r < 8; ++r) { mrow[r] = -3.0e38f; lrow[r] = 0.0f; }

  _Float16* myP = sP + wv * 16 * PS;

  // prologue: DMA tile 0 into buffer 0
  if (wv == 0) tdm_issue(xhb, (unsigned)(size_t)&sK[0][0]);

  int buf = 0;
  for (int kt = 0; kt < NTILES; ++kt) {
    __syncthreads();  // all waves done with sVt and with sK[buf^1] (tile kt-1)

    if (wv == 0) {
      if (kt + 1 < NTILES) {
        tdm_issue(xhb + (size_t)(kt + 1) * KVTILE * D_,
                  (unsigned)(size_t)&sK[buf ^ 1][0]);
        __builtin_amdgcn_s_wait_tensorcnt(1);  // in-order: tile kt complete
      } else {
        __builtin_amdgcn_s_wait_tensorcnt(0);
      }
    }

    // build transposed V tile from L2-resident f16 source (overlaps TDM wait)
    const _Float16* src = xhb + (size_t)kt * KVTILE * D_;
#pragma unroll
    for (int it = 0; it < 8; ++it) {
      const int idx = it * 256 + tid;   // v4h units, 2048 total
      const int row = idx >> 6;
      const int c4  = idx & 63;
      v4h h = *(const v4h*)(src + row * D_ + c4 * 4);
      sVt[(c4 * 4 + 0) * VTS + row] = h[0];
      sVt[(c4 * 4 + 1) * VTS + row] = h[1];
      sVt[(c4 * 4 + 2) * VTS + row] = h[2];
      sVt[(c4 * 4 + 3) * VTS + row] = h[3];
    }
    __syncthreads();  // publishes sVt and TDM completion of sK[buf]

    const _Float16* sKb = &sK[buf][0];

    // ---- S = Q K^T : two 16x16 score tiles ----
    v8f S0 = {}, S1 = {};
#pragma unroll
    for (int c = 0; c < 8; ++c) {
      const _Float16* k0 = sKb + l16 * KTS        + 32 * c + 16 * hh;
      const _Float16* k1 = sKb + (16 + l16) * KTS + 32 * c + 16 * hh;
      v16h Bf0 = cat8(*(const v8h*)k0, *(const v8h*)(k0 + 8));
      v16h Bf1 = cat8(*(const v8h*)k1, *(const v8h*)(k1 + 8));
      S0 = __builtin_amdgcn_wmma_f32_16x16x32_f16(false, Qf[c], false, Bf0,
                                                  (short)0, S0, false, false);
      S1 = __builtin_amdgcn_wmma_f32_16x16x32_f16(false, Qf[c], false, Bf1,
                                                  (short)0, S1, false, false);
    }

    // ---- online softmax in exp2 domain (row = r + 8*hh) ----
    float alpha[8];
#pragma unroll
    for (int r = 0; r < 8; ++r) {
      float v = fmaxf(S0[r], S1[r]);
#pragma unroll
      for (int off = 1; off < 16; off <<= 1) v = fmaxf(v, __shfl_xor(v, off, 32));
      const float mn = fmaxf(mrow[r], v);
      const float a  = exp2f(mrow[r] - mn);
      const float p0 = exp2f(S0[r] - mn);
      const float p1 = exp2f(S1[r] - mn);
      mrow[r] = mn;
      S0[r] = p0; S1[r] = p1;
      float rs = p0 + p1;
#pragma unroll
      for (int off = 1; off < 16; off <<= 1) rs += __shfl_xor(rs, off, 32);
      lrow[r] = lrow[r] * a + rs;
      alpha[r] = a;
    }

#pragma unroll
    for (int c = 0; c < 16; ++c)
#pragma unroll
      for (int r = 0; r < 8; ++r) O[c][r] *= alpha[r];

    // P -> per-wave LDS scratch (C-layout store, A-layout reload; LDS in-order per wave)
#pragma unroll
    for (int r = 0; r < 8; ++r) {
      myP[(r + 8 * hh) * PS + l16]      = (_Float16)S0[r];
      myP[(r + 8 * hh) * PS + 16 + l16] = (_Float16)S1[r];
    }
    v16h Pf = cat8(*(const v8h*)(myP + l16 * PS + 16 * hh),
                   *(const v8h*)(myP + l16 * PS + 16 * hh + 8));

    // ---- O += P V ----
#pragma unroll
    for (int c = 0; c < 16; ++c) {
      const _Float16* vp = sVt + (16 * c + l16) * VTS + 16 * hh;
      v16h Vf = cat8(*(const v8h*)vp, *(const v8h*)(vp + 8));
      O[c] = __builtin_amdgcn_wmma_f32_16x16x32_f16(false, Pf, false, Vf,
                                                    (short)0, O[c], false, false);
    }
    buf ^= 1;
  }

  // ---- normalize and store fp32 output ----
#pragma unroll
  for (int r = 0; r < 8; ++r) {
    const float rinv = 1.0f / lrow[r];
    float* op = out + xb + (size_t)(qrow0 + r + 8 * hh) * D_;
#pragma unroll
    for (int c = 0; c < 16; ++c) op[16 * c + l16] = O[c][r] * rinv;
  }
}

// ---------------- fallback: round-1 fp32-source kernel (no workspace needed) --------
__global__ __launch_bounds__(256, 1)
void fa_wmma_f32_kernel(const float* __restrict__ x, float* __restrict__ out) {
  __shared__ _Float16 sK [KVTILE * KTS];
  __shared__ _Float16 sVt[D_ * VTS];
  __shared__ _Float16 sP [8 * 16 * PS];

  const int tid  = threadIdx.x;
  const int lane = tid & 31;
  const int wv   = tid >> 5;
  const int hh   = lane >> 4;
  const int l16  = lane & 15;

  const int    b     = blockIdx.x / (N_ / QTILE);
  const int    qt    = blockIdx.x % (N_ / QTILE);
  const size_t xb    = (size_t)b * N_ * D_;
  const int    qrow0 = qt * QTILE + wv * 16;

  v16h Qf[8];
  {
    const float* qp = x + xb + (size_t)(qrow0 + l16) * D_;
#pragma unroll
    for (int c = 0; c < 8; ++c) {
      const int d0 = 32 * c + 16 * hh;
      v4f f0 = *(const v4f*)(qp + d0);
      v4f f1 = *(const v4f*)(qp + d0 + 4);
      v4f f2 = *(const v4f*)(qp + d0 + 8);
      v4f f3 = *(const v4f*)(qp + d0 + 12);
      v16h q;
#pragma unroll
      for (int i = 0; i < 4; ++i) {
        q[i]      = (_Float16)(f0[i] * QSCALE);
        q[4 + i]  = (_Float16)(f1[i] * QSCALE);
        q[8 + i]  = (_Float16)(f2[i] * QSCALE);
        q[12 + i] = (_Float16)(f3[i] * QSCALE);
      }
      Qf[c] = q;
    }
  }

  v8f O[16];
#pragma unroll
  for (int c = 0; c < 16; ++c) { v8f z = {}; O[c] = z; }
  float mrow[8], lrow[8];
#pragma unroll
  for (int r = 0; r < 8; ++r) { mrow[r] = -3.0e38f; lrow[r] = 0.0f; }

  _Float16* myP = sP + wv * 16 * PS;

  for (int kt = 0; kt < NTILES; ++kt) {
    const int kv0 = kt * KVTILE;
    __syncthreads();
#pragma unroll
    for (int it = 0; it < 8; ++it) {
      const int idx = it * 256 + tid;
      const int row = idx >> 6;
      const int c4  = idx & 63;
      v4f f = *(const v4f*)(x + xb + (size_t)(kv0 + row) * D_ + c4 * 4);
      _Float16 h0 = (_Float16)f[0], h1 = (_Float16)f[1];
      _Float16 h2 = (_Float16)f[2], h3 = (_Float16)f[3];
      *(v4h*)(sK + row * KTS + c4 * 4) = (v4h){h0, h1, h2, h3};
      sVt[(c4 * 4 + 0) * VTS + row] = h0;
      sVt[(c4 * 4 + 1) * VTS + row] = h1;
      sVt[(c4 * 4 + 2) * VTS + row] = h2;
      sVt[(c4 * 4 + 3) * VTS + row] = h3;
    }
    __syncthreads();

    v8f S0 = {}, S1 = {};
#pragma unroll
    for (int c = 0; c < 8; ++c) {
      const _Float16* k0 = sK + l16 * KTS        + 32 * c + 16 * hh;
      const _Float16* k1 = sK + (16 + l16) * KTS + 32 * c + 16 * hh;
      v16h Bf0 = cat8(*(const v8h*)k0, *(const v8h*)(k0 + 8));
      v16h Bf1 = cat8(*(const v8h*)k1, *(const v8h*)(k1 + 8));
      S0 = __builtin_amdgcn_wmma_f32_16x16x32_f16(false, Qf[c], false, Bf0,
                                                  (short)0, S0, false, false);
      S1 = __builtin_amdgcn_wmma_f32_16x16x32_f16(false, Qf[c], false, Bf1,
                                                  (short)0, S1, false, false);
    }

    float alpha[8];
#pragma unroll
    for (int r = 0; r < 8; ++r) {
      float v = fmaxf(S0[r], S1[r]);
#pragma unroll
      for (int off = 1; off < 16; off <<= 1) v = fmaxf(v, __shfl_xor(v, off, 32));
      const float mn = fmaxf(mrow[r], v);
      const float a  = exp2f(mrow[r] - mn);
      const float p0 = exp2f(S0[r] - mn);
      const float p1 = exp2f(S1[r] - mn);
      mrow[r] = mn;
      S0[r] = p0; S1[r] = p1;
      float rs = p0 + p1;
#pragma unroll
      for (int off = 1; off < 16; off <<= 1) rs += __shfl_xor(rs, off, 32);
      lrow[r] = lrow[r] * a + rs;
      alpha[r] = a;
    }

#pragma unroll
    for (int c = 0; c < 16; ++c)
#pragma unroll
      for (int r = 0; r < 8; ++r) O[c][r] *= alpha[r];

#pragma unroll
    for (int r = 0; r < 8; ++r) {
      myP[(r + 8 * hh) * PS + l16]      = (_Float16)S0[r];
      myP[(r + 8 * hh) * PS + 16 + l16] = (_Float16)S1[r];
    }
    v16h Pf = cat8(*(const v8h*)(myP + l16 * PS + 16 * hh),
                   *(const v8h*)(myP + l16 * PS + 16 * hh + 8));

#pragma unroll
    for (int c = 0; c < 16; ++c) {
      const _Float16* vp = sVt + (16 * c + l16) * VTS + 16 * hh;
      v16h Vf = cat8(*(const v8h*)vp, *(const v8h*)(vp + 8));
      O[c] = __builtin_amdgcn_wmma_f32_16x16x32_f16(false, Pf, false, Vf,
                                                    (short)0, O[c], false, false);
    }
  }

#pragma unroll
  for (int r = 0; r < 8; ++r) {
    const float rinv = 1.0f / lrow[r];
    float* op = out + xb + (size_t)(qrow0 + r + 8 * hh) * D_;
#pragma unroll
    for (int c = 0; c < 16; ++c) op[16 * c + l16] = O[c][r] * rinv;
  }
}

extern "C" void kernel_launch(void* const* d_in, const int* in_sizes, int n_in,
                              void* d_out, int out_size, void* d_ws, size_t ws_size,
                              hipStream_t stream) {
  const float* x   = (const float*)d_in[0];
  float*       out = (float*)d_out;
  (void)in_sizes; (void)n_in; (void)out_size;

  const size_t need = (size_t)B_ * N_ * D_ * sizeof(_Float16);  // 8 MB
  dim3 grid(B_ * (N_ / QTILE));   // 128 blocks
  dim3 block(256);                // 8 wave32 waves

  if (ws_size >= need && d_ws != nullptr) {
    _Float16* xh = (_Float16*)d_ws;
    hipLaunchKernelGGL(cvt_f16_kernel, dim3((B_ * N_ * D_) / 2048), dim3(256), 0,
                       stream, x, xh);
    hipLaunchKernelGGL(fa_wmma_tdm_kernel, grid, block, 0, stream, xh, out);
  } else {
    hipLaunchKernelGGL(fa_wmma_f32_kernel, grid, block, 0, stream, x, out);
  }
}
